// GraphEmbedding_66090956751483
// MI455X (gfx1250) — matrix operand order repaired and checked by reading, not compiled, not run on hardware
//
#include <hip/hip_runtime.h>
#include <math.h>

// ---------------------------------------------------------------------------
// WeaveNet graph embedding for MI455X (gfx1250, wave32, WMMA).
//   * all GEMMs via v_wmma_f32_16x16x32_f16 (f16 in, f32 accumulate)
//   * all intermediate activations stored f16 (halves HBM traffic; tanh-bounded)
//   * weights pre-converted once per launch to transposed zero-padded f16
//     (Wt[N][KR]) -> B fragments are direct global_load_b128 (L2-resident)
//   * layer-2 pair outputs are dead code in the reference -> skipped
//   * gather-GEMM fuses atom[atom_to_pair] concat into A staging
//   * segment_sum uses sorted pair_split: binary search + coalesced reduce
// Workspace: ~357 MB.
// ---------------------------------------------------------------------------

typedef __attribute__((ext_vector_type(16))) _Float16 v16h;
typedef __attribute__((ext_vector_type(8)))  _Float16 v8h;
typedef __attribute__((ext_vector_type(8)))  float    v8f;

#define N_ATOMS 20000
#define N_PAIRS 800000
#define ATOM_F  75
#define PAIR_F  14

#define BM 64
#define BN 64
#define BK 32
#define LDSW 48   // halfs per LDS row: 96B, 16B-aligned, conflict-reduced

// activation buffer row strides (halfs, multiples of 8 for aligned v8h loads)
#define ATOMCAT_LD 336   // cols: [0,75) input | [75,203) atom1 | [203,331) atom2
#define PAIRCAT_LD 80    // cols: [0,14) input | [14,78) pair1
#define PAIRTMP_LD 128
#define ATOMTMP_LD 256

// C (cols [colOff,colOff+N)) = tanh(A[M,K] @ Wt^T + bias); FINAL adds BN, f32 out.
// GATHER: A-row p = concat(Abase[gidx[2p]][0:aK], Abase[gidx[2p+1]][0:aK]), K=2*aK.
template<bool GATHER, bool FINAL>
__global__ __launch_bounds__(128)
void gemm_tanh_wmma(const _Float16* __restrict__ A, int lda, int M, int K,
                    const int* __restrict__ gidx, int aK,
                    const _Float16* __restrict__ Wt, int KR,   // Wt[N][KR], KR=ceil32(K), zero padded
                    const float* __restrict__ bias, int N,
                    void* __restrict__ Cv, int ldc, int colOff,
                    const float* __restrict__ bng, const float* __restrict__ bnb,
                    const float* __restrict__ bnm, const float* __restrict__ bnv)
{
    __shared__ __attribute__((aligned(16))) _Float16 As[BM * LDSW];

    const int tid  = threadIdx.x;
    const int lane = tid & 31;
    const int wave = tid >> 5;
    const long bm  = (long)blockIdx.x * BM;
    const int  bn  = blockIdx.y * BN;
    const int  r   = lane & 15;   // A row-in-16 / B col / C col
    const int  h   = lane >> 4;   // lane half -> K sub-range / C row block

    v8f acc[4] = {};

    for (int k0 = 0; k0 < K; k0 += BK) {
        // ---- stage A tile (64 rows x 32 k) into LDS in 8-half chunks ----
        #pragma unroll
        for (int i = 0; i < 2; ++i) {
            const int chunk = tid + i * 128;      // 0..255 = 64 rows x 4 chunks
            const int row = chunk >> 2;
            const int kc  = (chunk & 3) << 3;     // 0,8,16,24
            const int gk  = k0 + kc;
            const long grow = bm + row;
            v8h v = {0, 0, 0, 0, 0, 0, 0, 0};
            if (grow < M) {
                if constexpr (!GATHER) {
                    const _Float16* src = A + grow * (long)lda + gk;
                    if (gk + 8 <= K) {
                        v = *reinterpret_cast<const v8h*>(src);
                        if (gk + BK + 8 <= K)                 // speculative next-tile prefetch
                            __builtin_prefetch(src + BK, 0, 0);
                    } else if (gk < K) {
                        #pragma unroll
                        for (int e = 0; e < 8; ++e)
                            if (gk + e < K) v[e] = src[e];
                    }
                } else {
                    const int p0 = gidx[grow * 2 + 0];
                    const int p1 = gidx[grow * 2 + 1];
                    if (gk + 8 <= aK) {                       // fully inside atom_i slice
                        v = *reinterpret_cast<const v8h*>(A + (long)p0 * lda + gk);
                    } else {                                  // seam / atom_j slice (unaligned)
                        #pragma unroll
                        for (int e = 0; e < 8; ++e) {
                            const int g = gk + e;
                            if (g < aK)      v[e] = A[(long)p0 * lda + g];
                            else if (g < K)  v[e] = A[(long)p1 * lda + (g - aK)];
                        }
                    }
                }
            }
            *reinterpret_cast<v8h*>(&As[row * LDSW + kc]) = v;
        }
        __syncthreads();

        // ---- fragments: A from LDS (2x ds_load_b128), B direct from global Wt ----
        {
            const _Float16* arow = As + (wave * 16 + r) * LDSW;
            union { v16h v; v8h q[2]; } af;
            af.q[0] = *reinterpret_cast<const v8h*>(arow + 8 * h);       // k = 8h..8h+7
            af.q[1] = *reinterpret_cast<const v8h*>(arow + 16 + 8 * h);  // k = 16+8h..
            #pragma unroll
            for (int j = 0; j < 4; ++j) {
                const _Float16* wt = Wt + (size_t)(bn + j * 16 + r) * KR + k0 + 8 * h;
                union { v16h v; v8h q[2]; } bf;
                bf.q[0] = *reinterpret_cast<const v8h*>(wt);       // k = 8h..8h+7
                bf.q[1] = *reinterpret_cast<const v8h*>(wt + 16);  // k = 16+8h..
                acc[j] = __builtin_amdgcn_wmma_f32_16x16x32_f16(
                    false, af.v, false, bf.v, (short)0, acc[j], false, false);
            }
        }
        __syncthreads();
    }

    // ---- epilogue: bias + tanh (+ BN for FINAL). C layout: col = r, row = rr + 8h ----
    #pragma unroll
    for (int j = 0; j < 4; ++j) {
        const int gc = bn + j * 16 + r;
        const float bv = bias[gc];
        float scale = 1.f, shift = 0.f;
        if constexpr (FINAL) {
            const float iv = bng[gc] * rsqrtf(bnv[gc] + 1e-3f);
            scale = iv;
            shift = bnb[gc] - bnm[gc] * iv;
        }
        #pragma unroll
        for (int rr = 0; rr < 8; ++rr) {
            const long grow = bm + wave * 16 + rr + 8 * h;
            if (grow < M) {
                const float v = tanhf(acc[j][rr] + bv);
                if constexpr (FINAL)
                    ((float*)Cv)[grow * (long)ldc + colOff + gc] = v * scale + shift;
                else
                    ((_Float16*)Cv)[grow * (long)ldc + colOff + gc] = (_Float16)v;
            }
        }
    }
}

// W[K,N] fp32 -> Wt[N,KR] f16 transposed, zero-padded in K
__global__ void convert_wt(const float* __restrict__ W, int K, int N, int KR,
                           _Float16* __restrict__ Wt)
{
    const long i = blockIdx.x * (long)blockDim.x + threadIdx.x;
    if (i >= (long)N * KR) return;
    const int n = (int)(i / KR);
    const int k = (int)(i - (long)n * KR);
    Wt[i] = (_Float16)((k < K) ? W[(long)k * N + n] : 0.f);
}

// out[a, colOff+c] = sum_{p: seg[p]==a} PA[p,c]  (seg sorted ascending), f32 accumulate
__global__ __launch_bounds__(128)
void segsum_kernel(const _Float16* __restrict__ PA, const int* __restrict__ seg,
                   int nPairs, int F, _Float16* __restrict__ out, int ldo, int colOff)
{
    const int a = blockIdx.x;
    const int c = threadIdx.x;   // blockDim.x == F
    int lo = 0, hi = nPairs;
    while (lo < hi) { const int mid = (lo + hi) >> 1; if (seg[mid] <  a) lo = mid + 1; else hi = mid; }
    const int start = lo;
    hi = nPairs;
    while (lo < hi) { const int mid = (lo + hi) >> 1; if (seg[mid] <= a) lo = mid + 1; else hi = mid; }
    const int end = lo;
    float s = 0.f;
    for (int p = start; p < end; ++p) s += (float)PA[(long)p * F + c];
    out[(long)a * ldo + colOff + c] = (_Float16)s;
}

// dst[row, c] = (f16)src[row, c]   (src dense [rows, srcCols], dst stride ldd)
__global__ void copy_cast(const float* __restrict__ src, int srcCols,
                          _Float16* __restrict__ dst, int ldd, long n)
{
    const long i = blockIdx.x * (long)blockDim.x + threadIdx.x;
    if (i >= n) return;
    const long row = i / srcCols;
    const int  c   = (int)(i - row * srcCols);
    dst[row * (long)ldd + c] = (_Float16)src[i];
}

static inline char* carve(char*& p, size_t bytes) {
    char* r = p;
    p += (bytes + 255) & ~(size_t)255;
    return r;
}

static inline _Float16* conv_w(hipStream_t s, char*& p, const float* W, int K, int N, int* KRout) {
    const int KR = (K + 31) & ~31;
    _Float16* wt = (_Float16*)carve(p, (size_t)N * KR * sizeof(_Float16));
    const long n = (long)N * KR;
    convert_wt<<<(unsigned)((n + 255) / 256), 256, 0, s>>>(W, K, N, KR, wt);
    *KRout = KR;
    return wt;
}

static inline void gemm(hipStream_t s, int mode,  // 0=plain, 1=gather, 2=final(BN,f32)
                        const _Float16* A, int lda, int M, int K,
                        const int* gidx, int aK,
                        const _Float16* Wt, int KR, const float* b, int N,
                        void* C, int ldc, int colOff,
                        const float* g = nullptr, const float* be = nullptr,
                        const float* mu = nullptr, const float* va = nullptr)
{
    dim3 grid((unsigned)((M + BM - 1) / BM), (unsigned)(N / BN));
    if (mode == 1)
        gemm_tanh_wmma<true,  false><<<grid, 128, 0, s>>>(A, lda, M, K, gidx, aK, Wt, KR, b, N,
                                                          C, ldc, colOff, nullptr, nullptr, nullptr, nullptr);
    else if (mode == 2)
        gemm_tanh_wmma<false, true ><<<grid, 128, 0, s>>>(A, lda, M, K, nullptr, 0, Wt, KR, b, N,
                                                          C, ldc, colOff, g, be, mu, va);
    else
        gemm_tanh_wmma<false, false><<<grid, 128, 0, s>>>(A, lda, M, K, nullptr, 0, Wt, KR, b, N,
                                                          C, ldc, colOff, nullptr, nullptr, nullptr, nullptr);
}

extern "C" void kernel_launch(void* const* d_in, const int* in_sizes, int n_in,
                              void* d_out, int out_size, void* d_ws, size_t ws_size,
                              hipStream_t stream)
{
    (void)out_size; (void)ws_size;
    // Flattening order detection: insertion order => d_in[1] is pair_features (800000*14),
    // jax tree_leaves (sorted dict keys) => d_in[1] is atom_split (20000).
    const bool insertion = (n_in > 1 && in_sizes[1] == N_PAIRS * PAIR_F);

    const float *atom_features, *pair_features;
    const int *pair_split, *atom_to_pair;
    const float *W_AA[2], *b_AA[2], *W_PA[2], *b_PA[2], *W_A[2], *b_A[2],
                *W_AP[2], *b_AP[2], *W_PP[2], *b_PP[2], *W_P[2], *b_P[2];
    const float *dense_W, *dense_b, *bng, *bnb, *bnm, *bnv;

    if (insertion) {
        atom_features = (const float*)d_in[0];
        pair_features = (const float*)d_in[1];
        pair_split    = (const int*)d_in[2];
        atom_to_pair  = (const int*)d_in[4];
        int base = 5;
        for (int l = 0; l < 2; ++l) {
            W_AA[l] = (const float*)d_in[base + 0];  b_AA[l] = (const float*)d_in[base + 1];
            W_PA[l] = (const float*)d_in[base + 2];  b_PA[l] = (const float*)d_in[base + 3];
            W_A [l] = (const float*)d_in[base + 4];  b_A [l] = (const float*)d_in[base + 5];
            W_AP[l] = (const float*)d_in[base + 6];  b_AP[l] = (const float*)d_in[base + 7];
            W_PP[l] = (const float*)d_in[base + 8];  b_PP[l] = (const float*)d_in[base + 9];
            W_P [l] = (const float*)d_in[base +10];  b_P [l] = (const float*)d_in[base +11];
            base += 12;
        }
        dense_W = (const float*)d_in[29]; dense_b = (const float*)d_in[30];
        bng = (const float*)d_in[31]; bnb = (const float*)d_in[32];
        bnm = (const float*)d_in[33]; bnv = (const float*)d_in[34];
    } else {
        atom_features = (const float*)d_in[0];
        atom_to_pair  = (const int*)d_in[2];
        pair_features = (const float*)d_in[3];
        pair_split    = (const int*)d_in[4];
        bnb = (const float*)d_in[5]; bng = (const float*)d_in[6];
        bnm = (const float*)d_in[7]; bnv = (const float*)d_in[8];
        dense_W = (const float*)d_in[9]; dense_b = (const float*)d_in[10];
        int base = 11;
        for (int l = 0; l < 2; ++l) {
            W_A [l] = (const float*)d_in[base + 0];  W_AA[l] = (const float*)d_in[base + 1];
            W_AP[l] = (const float*)d_in[base + 2];  W_P [l] = (const float*)d_in[base + 3];
            W_PA[l] = (const float*)d_in[base + 4];  W_PP[l] = (const float*)d_in[base + 5];
            b_A [l] = (const float*)d_in[base + 6];  b_AA[l] = (const float*)d_in[base + 7];
            b_AP[l] = (const float*)d_in[base + 8];  b_P [l] = (const float*)d_in[base + 9];
            b_PA[l] = (const float*)d_in[base +10];  b_PP[l] = (const float*)d_in[base +11];
            base += 12;
        }
    }

    // ---- workspace (f16 activations) ----
    char* p = (char*)d_ws;
    _Float16* atomcat = (_Float16*)carve(p, (size_t)N_ATOMS * ATOMCAT_LD * 2);
    _Float16* paircat = (_Float16*)carve(p, (size_t)N_PAIRS * PAIRCAT_LD * 2);
    _Float16* pairtmp = (_Float16*)carve(p, (size_t)N_PAIRS * PAIRTMP_LD * 2);
    _Float16* atomtmp = (_Float16*)carve(p, (size_t)N_ATOMS * ATOMTMP_LD * 2);

    // ---- one-time weight convert/transpose/pad to f16 (L2-resident afterwards) ----
    int krAA1, krPA1, krA1, krAP1, krPP1, krP1, krAA2, krPA2, krA2, krD;
    _Float16* wtAA1 = conv_w(stream, p, W_AA[0],  75, 128, &krAA1);
    _Float16* wtPA1 = conv_w(stream, p, W_PA[0],  14, 128, &krPA1);
    _Float16* wtA1  = conv_w(stream, p, W_A [0], 256, 128, &krA1);
    _Float16* wtAP1 = conv_w(stream, p, W_AP[0], 150,  64, &krAP1);
    _Float16* wtPP1 = conv_w(stream, p, W_PP[0],  14,  64, &krPP1);
    _Float16* wtP1  = conv_w(stream, p, W_P [0], 128,  64, &krP1);
    _Float16* wtAA2 = conv_w(stream, p, W_AA[1], 203, 128, &krAA2);
    _Float16* wtPA2 = conv_w(stream, p, W_PA[1],  78, 128, &krPA2);
    _Float16* wtA2  = conv_w(stream, p, W_A [1], 256, 128, &krA2);
    _Float16* wtD   = conv_w(stream, p, dense_W, 331, 256, &krD);

    // ---- input copies (fp32 -> f16 concat buffers) ----
    {
        long n = (long)N_ATOMS * ATOM_F;
        copy_cast<<<(unsigned)((n + 255) / 256), 256, 0, stream>>>(atom_features, ATOM_F, atomcat, ATOMCAT_LD, n);
        n = (long)N_PAIRS * PAIR_F;
        copy_cast<<<(unsigned)((n + 255) / 256), 256, 0, stream>>>(pair_features, PAIR_F, paircat, PAIRCAT_LD, n);
    }

    // ---- weave layer 1 (a_in=75, p_in=14, ah=128, ph=64) ----
    gemm(stream, 0, atomcat, ATOMCAT_LD, N_ATOMS, 75,  nullptr, 0, wtAA1, krAA1, b_AA[0], 128, atomtmp, ATOMTMP_LD, 0);
    gemm(stream, 0, paircat, PAIRCAT_LD, N_PAIRS, 14,  nullptr, 0, wtPA1, krPA1, b_PA[0], 128, pairtmp, PAIRTMP_LD, 0);
    segsum_kernel<<<N_ATOMS, 128, 0, stream>>>(pairtmp, pair_split, N_PAIRS, PAIRTMP_LD, atomtmp, ATOMTMP_LD, 128);
    gemm(stream, 0, atomtmp, ATOMTMP_LD, N_ATOMS, 256, nullptr, 0, wtA1,  krA1,  b_A [0], 128, atomcat, ATOMCAT_LD, 75);
    gemm(stream, 1, atomcat, ATOMCAT_LD, N_PAIRS, 150, atom_to_pair, 75, wtAP1, krAP1, b_AP[0], 64, pairtmp, PAIRTMP_LD, 0);
    gemm(stream, 0, paircat, PAIRCAT_LD, N_PAIRS, 14,  nullptr, 0, wtPP1, krPP1, b_PP[0], 64,  pairtmp, PAIRTMP_LD, 64);
    gemm(stream, 0, pairtmp, PAIRTMP_LD, N_PAIRS, 128, nullptr, 0, wtP1,  krP1,  b_P [0], 64,  paircat, PAIRCAT_LD, 14);

    // ---- weave layer 2 (a_in=203, p_in=78); pair-side outputs are dead -> skipped ----
    gemm(stream, 0, atomcat, ATOMCAT_LD, N_ATOMS, 203, nullptr, 0, wtAA2, krAA2, b_AA[1], 128, atomtmp, ATOMTMP_LD, 0);
    gemm(stream, 0, paircat, PAIRCAT_LD, N_PAIRS, 78,  nullptr, 0, wtPA2, krPA2, b_PA[1], 128, pairtmp, PAIRTMP_LD, 0);
    segsum_kernel<<<N_ATOMS, 128, 0, stream>>>(pairtmp, pair_split, N_PAIRS, PAIRTMP_LD, atomtmp, ATOMTMP_LD, 128);
    gemm(stream, 0, atomtmp, ATOMTMP_LD, N_ATOMS, 256, nullptr, 0, wtA2,  krA2,  b_A [1], 128, atomcat, ATOMCAT_LD, 203);

    // ---- dense head + fused inference BatchNorm, straight into d_out (fp32) ----
    gemm(stream, 2, atomcat, ATOMCAT_LD, N_ATOMS, 331, nullptr, 0, wtD, krD, dense_b, 256,
         (float*)d_out, 256, 0, bng, bnb, bnm, bnv);
}
